// AttentionBlock_8538394985182
// MI455X (gfx1250) — matrix operand order repaired and checked by reading, not compiled, not run on hardware
//
#include <hip/hip_runtime.h>
#include <cmath>
#include <cstdint>

#define DIMM 768
#define SEQ  2048
#define NBAT 2
#define NHEAD 12
#define HDIM (NHEAD * DIMM)   /* 9216 */
#define DFF  (4 * DIMM)       /* 3072 */

typedef __bf16 bf16;
typedef __attribute__((ext_vector_type(16))) __bf16 bf16x16;
typedef __attribute__((ext_vector_type(8)))  float  f32x8;

struct __attribute__((aligned(16))) V128 { unsigned int x, y, z, w; };

union FragAB { V128 q[2]; bf16x16 v; };
union Pack8  { V128 q; bf16 h[8]; };

__device__ __forceinline__ bf16 f2bf(float f) {
  union { float f; unsigned int u; } c; c.f = f;
  unsigned int r = c.u + 0x7FFFu + ((c.u >> 16) & 1u);
  union { unsigned short u; bf16 h; } o; o.u = (unsigned short)(r >> 16);
  return o.h;
}

// LDS byte offset of a __shared__ object (generic -> AS3 addrspacecast -> int)
__device__ __forceinline__ unsigned lds_addr_of(const void* p) {
  return (unsigned)(unsigned long long)(__attribute__((address_space(3))) const char*)p;
}

// Async global->LDS 16-byte DMA (per active lane), tracked by ASYNCcnt.
__device__ __forceinline__ void async_g2l_b128(unsigned lds_byte_off,
                                               const bf16* gptr) {
  asm volatile("global_load_async_to_lds_b128 %0, %1, off"
               :
               : "v"(lds_byte_off),
                 "v"((unsigned long long)(size_t)gptr)
               : "memory");
}

// ---------------------------------------------------------------- cast
__global__ void cast_f32_bf16(const float* __restrict__ in,
                              bf16* __restrict__ out, int n) {
  int i = blockIdx.x * 256 + threadIdx.x;
  if (i < n) out[i] = f2bf(in[i]);
}

// ---------------------------------------------------------------- GEMM
// C[M,N] = A[M,K] @ op(B);  transB==0 : B is [K,N] (ld=N)
//                           transB==1 : B is [N,K] (ld=K)  (computes A @ B^T)
// epi: 0 -> f32 out, 1 -> bf16 out, 2 -> exact-GELU -> bf16, 3 -> +resid -> f32
// Double-buffered LDS; A (and B when transB) staged with async global->LDS DMA.
__global__ __launch_bounds__(256)
void gemm_wmma(const bf16* __restrict__ A, long long aStride, int aDiv,
               const bf16* __restrict__ Bm, long long bStride, int bMod,
               void* __restrict__ Cout, long long cStride,
               const float* __restrict__ resid,
               int M, int N, int K, int ldA, int ldB,
               int transB, int epi) {
  __shared__ __attribute__((aligned(16))) bf16 aT[2][128 * 40];
  __shared__ __attribute__((aligned(16))) bf16 bT[2][128 * 40];

  const int tid  = threadIdx.x;
  const int lane = tid & 31;
  const int wave = tid >> 5;
  const int wm   = wave >> 1;   // 0..3 -> 32-row strip
  const int wn   = wave & 1;    // 0..1 -> 64-col strip
  const int z    = blockIdx.z;
  const int m0   = blockIdx.y * 128;
  const int n0   = blockIdx.x * 128;

  const bf16* Ag = A  + (long long)(z / aDiv) * aStride;
  const bf16* Bg = Bm + (long long)(z % bMod) * bStride;

  f32x8 acc[2][4];
  for (int i = 0; i < 2; ++i)
    for (int j = 0; j < 4; ++j) acc[i][j] = (f32x8)0.0f;

  const int ln4 = lane & 15;
  const int k0a = (lane < 16) ? 0 : 8;    // A-frag K base (16-bit 16x32 layout)
  const int k0b = (lane < 16) ? 0 : 8;    // B-frag K base (16-bit 32x16 layout)
  const int mh  = (lane < 16) ? 0 : 8;    // C-frag row half

  auto stageA = [&](int buf, int k0) {
    unsigned base = lds_addr_of(&aT[buf][0]);
    for (int s = tid; s < 512; s += 256) {
      int row = s >> 2, c4 = (s & 3) * 8;
      async_g2l_b128(base + (unsigned)(row * 80 + c4 * 2),
                     Ag + (long long)(m0 + row) * ldA + k0 + c4);
    }
  };
  auto stageB = [&](int buf, int k0) {
    if (transB) {
      unsigned base = lds_addr_of(&bT[buf][0]);
      for (int s = tid; s < 512; s += 256) {
        int row = s >> 2, c4 = (s & 3) * 8;
        async_g2l_b128(base + (unsigned)(row * 80 + c4 * 2),
                       Bg + (long long)(n0 + row) * ldB + k0 + c4);
      }
    } else {
      bf16* bt = &bT[buf][0];
      for (int s = tid; s < 4096; s += 256) {
        int kk = s >> 7, nn = s & 127;
        bt[nn * 40 + kk] = Bg[(long long)(k0 + kk) * ldB + n0 + nn];
      }
    }
  };

  const int KT = K / 32;
  stageA(0, 0);
  stageB(0, 0);

  for (int kt = 0; kt < KT; ++kt) {
    const int buf = kt & 1;
    if (kt + 1 < KT) {
      // prefetch next tile into the other buffer while waiting only for
      // the current tile's async group (in-order completion)
      stageA(buf ^ 1, (kt + 1) * 32);
      stageB(buf ^ 1, (kt + 1) * 32);
      if (transB) asm volatile("s_wait_asynccnt 0x4" ::: "memory");
      else        asm volatile("s_wait_asynccnt 0x2" ::: "memory");
    } else {
      asm volatile("s_wait_asynccnt 0x0" ::: "memory");
    }
    __syncthreads();

    FragAB af[2], bfg[4];
    for (int i = 0; i < 2; ++i) {
      const bf16* p = &aT[buf][(wm * 32 + i * 16 + ln4) * 40 + k0a];
      af[i].q[0] = *(const V128*)p;
      af[i].q[1] = *(const V128*)(p + 16);
    }
    for (int j = 0; j < 4; ++j) {
      const bf16* p = &bT[buf][(wn * 64 + j * 16 + ln4) * 40 + k0b];
      bfg[j].q[0] = *(const V128*)p;
      bfg[j].q[1] = *(const V128*)(p + 16);
    }
    for (int i = 0; i < 2; ++i)
      for (int j = 0; j < 4; ++j)
        acc[i][j] = __builtin_amdgcn_wmma_f32_16x16x32_bf16(
            false, af[i].v, false, bfg[j].v, (short)0, acc[i][j], false, false);
    __syncthreads();
  }

  float* cF = (float*)Cout + (long long)z * cStride;
  bf16*  cB = (bf16*)Cout + (long long)z * cStride;
  for (int i = 0; i < 2; ++i)
    for (int j = 0; j < 4; ++j)
      for (int r = 0; r < 8; ++r) {
        int row = m0 + wm * 32 + i * 16 + r + mh;
        int col = n0 + wn * 64 + j * 16 + ln4;
        if (row < M && col < N) {
          float v = acc[i][j][r];
          long long idx = (long long)row * N + col;
          if (epi == 0)      cF[idx] = v;
          else if (epi == 1) cB[idx] = f2bf(v);
          else if (epi == 2) cB[idx] =
              f2bf(0.5f * v * (1.0f + erff(v * 0.70710678118654752f)));
          else               cF[idx] = v + resid[idx];
        }
      }
}

// ---------------------------------------------------------------- flash attention
// One block per (16 q-rows, head, batch). Online softmax, causal.
__global__ __launch_bounds__(256)
void flash_attn(const bf16* __restrict__ Q, const bf16* __restrict__ K,
                const bf16* __restrict__ V, bf16* __restrict__ ctx,
                float scale) {
  __shared__ __attribute__((aligned(16))) bf16 Vt[DIMM * 32];  // Vt[c][k]
  __shared__ float sc[16 * 32];
  __shared__ __attribute__((aligned(16))) bf16 pb[16 * 40];
  __shared__ float mrow[16], lrow[16], arow[16];

  const int tid  = threadIdx.x;
  const int lane = tid & 31;
  const int wave = tid >> 5;
  const int ln4  = lane & 15;
  const int q0   = blockIdx.x * 16;
  const int h    = blockIdx.y;
  const int b    = blockIdx.z;
  const long long base = ((long long)(b * NHEAD + h)) * SEQ * DIMM;

  f32x8 o[6];
  for (int j = 0; j < 6; ++j) o[j] = (f32x8)0.0f;
  const int colBase = wave * 96;

  if (tid < 16) { mrow[tid] = -1e30f; lrow[tid] = 0.0f; }
  __syncthreads();

  const int k0a = (lane < 16) ? 0 : 8;
  const int k0b = (lane < 16) ? 0 : 8;
  const int mh  = (lane < 16) ? 0 : 8;
  const int nkb = ((q0 + 15) >> 5) + 1;

  for (int kb = 0; kb < nkb; ++kb) {
    const int ks = kb * 32;
    // stage V tile transposed: Vt[c][k] = V[ks+k][c]
    for (int s = tid; s < 32 * (DIMM / 8); s += 256) {
      int k  = s / (DIMM / 8);
      int c8 = (s % (DIMM / 8)) * 8;
      Pack8 v; v.q = *(const V128*)(V + base + (long long)(ks + k) * DIMM + c8);
      for (int e = 0; e < 8; ++e) Vt[(c8 + e) * 32 + k] = v.h[e];
    }
    // scores: waves 0,1 each compute a 16x16 tile of Q·K^T (K-dim = 768)
    if (wave < 2) {
      f32x8 s8 = (f32x8)0.0f;
      const int key = ks + wave * 16 + ln4;
      const bf16* qrow = Q + base + (long long)(q0 + ln4) * DIMM;
      const bf16* krow = K + base + (long long)key * DIMM;
      for (int kc = 0; kc < DIMM; kc += 32) {
        FragAB a, bb;
        a.q[0]  = *(const V128*)(qrow + kc + k0a);
        a.q[1]  = *(const V128*)(qrow + kc + k0a + 16);
        bb.q[0] = *(const V128*)(krow + kc + k0b);
        bb.q[1] = *(const V128*)(krow + kc + k0b + 16);
        s8 = __builtin_amdgcn_wmma_f32_16x16x32_bf16(
            false, a.v, false, bb.v, (short)0, s8, false, false);
      }
      for (int r = 0; r < 8; ++r) {
        int row = r + mh;
        int j   = ks + wave * 16 + ln4;
        sc[row * 32 + wave * 16 + ln4] =
            (j <= q0 + row) ? s8[r] * scale : -1e30f;
      }
    }
    __syncthreads();
    // online softmax bookkeeping, one thread per q-row
    if (tid < 16) {
      const int row = tid;
      float m = mrow[row], bm = -1e30f;
      for (int j = 0; j < 32; ++j) bm = fmaxf(bm, sc[row * 32 + j]);
      float nm = fmaxf(m, bm);
      float al = __expf(m - nm);
      float sum = 0.0f;
      for (int j = 0; j < 32; ++j) {
        float p = __expf(sc[row * 32 + j] - nm);
        sum += p;
        pb[row * 40 + j] = f2bf(p);
      }
      mrow[row] = nm;
      lrow[row] = lrow[row] * al + sum;
      arow[row] = al;
    }
    __syncthreads();
    // rescale accumulators and do P·V (K-dim = 32 keys, one WMMA per tile)
    FragAB a;
    {
      const bf16* p = &pb[ln4 * 40 + k0a];
      a.q[0] = *(const V128*)p;
      a.q[1] = *(const V128*)(p + 16);
    }
    for (int j = 0; j < 6; ++j)
      for (int r = 0; r < 8; ++r) o[j][r] *= arow[r + mh];
    for (int j = 0; j < 6; ++j) {
      FragAB bb;
      const bf16* p = &Vt[(colBase + j * 16 + ln4) * 32 + k0b];
      bb.q[0] = *(const V128*)p;
      bb.q[1] = *(const V128*)(p + 16);
      o[j] = __builtin_amdgcn_wmma_f32_16x16x32_bf16(
          false, a.v, false, bb.v, (short)0, o[j], false, false);
    }
    __syncthreads();
  }
  // normalize and emit ctx in [B, S, H*D] layout (bf16 for the W_o GEMM)
  for (int j = 0; j < 6; ++j)
    for (int r = 0; r < 8; ++r) {
      int row = r + mh;
      int col = colBase + j * 16 + ln4;
      float v = o[j][r] / lrow[row];
      ctx[((long long)(b * SEQ + q0 + row)) * HDIM + h * DIMM + col] = f2bf(v);
    }
}

// ---------------------------------------------------------------- layernorm (no affine)
__global__ __launch_bounds__(256)
void layernorm(const float* __restrict__ in, bf16* __restrict__ out) {
  __shared__ float rs[256], rs2[256];
  const int row = blockIdx.x;
  const float* x = in + (long long)row * DIMM;
  float s = 0.0f, s2 = 0.0f;
  for (int i = threadIdx.x; i < DIMM; i += 256) {
    float v = x[i]; s += v; s2 += v * v;
  }
  rs[threadIdx.x] = s; rs2[threadIdx.x] = s2;
  __syncthreads();
  for (int st = 128; st > 0; st >>= 1) {
    if (threadIdx.x < st) {
      rs[threadIdx.x]  += rs[threadIdx.x + st];
      rs2[threadIdx.x] += rs2[threadIdx.x + st];
    }
    __syncthreads();
  }
  float mean = rs[0] / DIMM;
  float var  = rs2[0] / DIMM - mean * mean;
  float inv  = rsqrtf(var + 1e-5f);
  for (int i = threadIdx.x; i < DIMM; i += 256)
    out[(long long)row * DIMM + i] = f2bf((x[i] - mean) * inv);
}

// ---------------------------------------------------------------- launcher
extern "C" void kernel_launch(void* const* d_in, const int* in_sizes, int n_in,
                              void* d_out, int out_size, void* d_ws, size_t ws_size,
                              hipStream_t stream) {
  (void)in_sizes; (void)n_in; (void)out_size; (void)ws_size;
  const float* x  = (const float*)d_in[0];
  const float* Wq = (const float*)d_in[1];
  const float* Wk = (const float*)d_in[2];
  const float* Wv = (const float*)d_in[3];
  const float* Wo = (const float*)d_in[4];
  const float* W1 = (const float*)d_in[5];
  const float* W2 = (const float*)d_in[6];

  char* ws = (char*)d_ws;
  size_t off = 0;
  auto carve = [&](size_t bytes) -> void* {
    void* p = ws + off;
    off = (off + bytes + 255) & ~(size_t)255;
    return p;
  };

  const size_t nX   = (size_t)NBAT * SEQ * DIMM;
  const size_t nWq  = (size_t)NHEAD * DIMM * DIMM;
  const size_t nWo  = (size_t)DIMM * HDIM;
  const size_t nW1  = (size_t)DFF * DIMM;
  const size_t nQKV = (size_t)NBAT * NHEAD * SEQ * DIMM;
  const size_t nCtx = (size_t)NBAT * SEQ * HDIM;
  const size_t nHid = (size_t)NBAT * SEQ * DFF;

  bf16*  xb   = (bf16*)carve(nX * 2);
  bf16*  wqb  = (bf16*)carve(nWq * 2);
  bf16*  wkb  = (bf16*)carve(nWq * 2);
  bf16*  wvb  = (bf16*)carve(nWq * 2);
  bf16*  wob  = (bf16*)carve(nWo * 2);
  bf16*  w1b  = (bf16*)carve(nW1 * 2);
  bf16*  w2b  = (bf16*)carve(nW1 * 2);
  bf16*  Qb   = (bf16*)carve(nQKV * 2);
  bf16*  Kb   = (bf16*)carve(nQKV * 2);
  bf16*  Vb   = (bf16*)carve(nQKV * 2);
  bf16*  ctxb = (bf16*)carve(nCtx * 2);
  float* attn = (float*)carve(nX * 4);
  bf16*  lnb  = (bf16*)carve(nX * 2);
  bf16*  hb   = (bf16*)carve(nHid * 2);

  auto cast = [&](const float* src, bf16* dst, size_t n) {
    cast_f32_bf16<<<dim3((unsigned)((n + 255) / 256)), dim3(256), 0, stream>>>(
        src, dst, (int)n);
  };
  cast(x,  xb,  nX);
  cast(Wq, wqb, nWq);
  cast(Wk, wkb, nWq);
  cast(Wv, wvb, nWq);
  cast(Wo, wob, nWo);
  cast(W1, w1b, nW1);
  cast(W2, w2b, nW1);

  // QKV projections: per (b,h)  Q[b,h] = x[b] @ W[h]   (nn, K=N=768, M=2048)
  {
    dim3 g(DIMM / 128, SEQ / 128, NBAT * NHEAD);
    gemm_wmma<<<g, dim3(256), 0, stream>>>(
        xb, (long long)SEQ * DIMM, NHEAD, wqb, (long long)DIMM * DIMM, NHEAD,
        (void*)Qb, (long long)SEQ * DIMM, nullptr,
        SEQ, DIMM, DIMM, DIMM, DIMM, 0, 1);
    gemm_wmma<<<g, dim3(256), 0, stream>>>(
        xb, (long long)SEQ * DIMM, NHEAD, wkb, (long long)DIMM * DIMM, NHEAD,
        (void*)Kb, (long long)SEQ * DIMM, nullptr,
        SEQ, DIMM, DIMM, DIMM, DIMM, 0, 1);
    gemm_wmma<<<g, dim3(256), 0, stream>>>(
        xb, (long long)SEQ * DIMM, NHEAD, wvb, (long long)DIMM * DIMM, NHEAD,
        (void*)Vb, (long long)SEQ * DIMM, nullptr,
        SEQ, DIMM, DIMM, DIMM, DIMM, 0, 1);
  }

  // causal flash attention -> ctx [B, S, H*D] bf16
  flash_attn<<<dim3(SEQ / 16, NHEAD, NBAT), dim3(256), 0, stream>>>(
      Qb, Kb, Vb, ctxb, 1.0f / sqrtf((float)DIMM));

  // attn_out = ctx @ Wo^T   (nt, M=4096, N=768, K=9216) -> f32
  gemm_wmma<<<dim3(DIMM / 128, (NBAT * SEQ) / 128, 1), dim3(256), 0, stream>>>(
      ctxb, 0LL, 1, wob, 0LL, 1, (void*)attn, 0LL, nullptr,
      NBAT * SEQ, DIMM, HDIM, HDIM, HDIM, 1, 0);

  // LN (no affine) -> bf16
  layernorm<<<dim3(NBAT * SEQ), dim3(256), 0, stream>>>(attn, lnb);

  // h = GELU(ln @ W1^T)   (nt, M=4096, N=3072, K=768) -> bf16
  gemm_wmma<<<dim3(DFF / 128, (NBAT * SEQ) / 128, 1), dim3(256), 0, stream>>>(
      lnb, 0LL, 1, w1b, 0LL, 1, (void*)hb, 0LL, nullptr,
      NBAT * SEQ, DFF, DIMM, DIMM, DIMM, 1, 2);

  // out = attn_out + h @ W2^T   (nt, M=4096, N=768, K=3072) -> f32 d_out
  gemm_wmma<<<dim3(DIMM / 128, (NBAT * SEQ) / 128, 1), dim3(256), 0, stream>>>(
      hb, 0LL, 1, w2b, 0LL, 1, d_out, 0LL, attn,
      NBAT * SEQ, DIMM, DFF, DFF, DFF, 1, 3);
}
